// EdgeBeliefs_28381143892852
// MI455X (gfx1250) — matrix operand order repaired and checked by reading, not compiled, not run on hardware
//
#include <hip/hip_runtime.h>

// EdgeBeliefs on MI455X (gfx1250, wave32).
//
// Phase 0: sigP = sigmoid(P)                       (tiny elementwise kernel)
// Phase 1: sequential sampling of node order       (1 block / dag, 128 thr)
//          - sigP cached in LDS, rows padded to 129 floats (bank-conflict free)
//          - incremental row-max maintenance
//          - wave32 shfl reductions/scans for softmax + cumsum
// Phase 2: gumbel straight-through sample + un-permutation (1 block / dag)
//          - identity: logA - log1mA == P, so Y[a,b] = (P[oa,ob] + g0-g1 >= 0)
//          - Y stored TRANSPOSED + column-permuted in LDS:
//              sWt[order[a]][b] = Y[a,b]   (row stride 136 halves = 272 B)
//            so WMMA B fragments are 16 contiguous halves per lane
//            (2x ds_load_b128, conflict-free).
//          - out = Pm * W via v_wmma_f32_16x16x32_f16 (exact for 0/1 f16).
//          Memory-bound: ~200 MB/dispatch -> ~9 us at 23.3 TB/s HBM.

#define NODES 128
#define WSTR  136          // halves per LDS row: 272 B = 17*16 B (aligned, conflict-free)
#define NEGV  (-1e30f)

typedef __attribute__((ext_vector_type(16))) _Float16 v16h;
typedef __attribute__((ext_vector_type(8)))  _Float16 v8h;
typedef __attribute__((ext_vector_type(4)))  _Float16 v4h;
typedef __attribute__((ext_vector_type(8)))  float    v8f;
typedef __attribute__((ext_vector_type(4)))  float    v4f;

// ---------------------------------------------------------------- Phase 0
__global__ void eb_sigmoid_kernel(const float* __restrict__ P,
                                  float* __restrict__ sigP) {
  int i = blockIdx.x * blockDim.x + threadIdx.x;
  if (i < NODES * NODES) sigP[i] = 1.0f / (1.0f + __expf(-P[i]));
}

// ---------------------------------------------------------------- Phase 1
__global__ void __launch_bounds__(128)
eb_order_kernel(const float* __restrict__ sigP,
                const float* __restrict__ uniforms,
                int*   __restrict__ order_ws,
                int*   __restrict__ inv_ws,
                float* __restrict__ order_out) {
  extern __shared__ float lds[];
  float* rows     = lds;                          // 128 * 129 floats
  int*   s_order  = (int*)(lds + NODES * 129);    // 128
  int*   s_active = s_order + NODES;              // 128
  float* s_red    = (float*)(s_active + NODES);   // 4 (wave maxes)
  float* s_wsum   = s_red + 4;                    // 4 (wave scan totals)
  int*   s_sel    = (int*)(s_wsum + 4);           // [0]=min cond idx, [1]=max active

  const int d    = blockIdx.x;
  const int i    = threadIdx.x;                   // row owned by this thread
  const int lane = i & 31;
  const int wid  = i >> 5;

  for (int idx = i; idx < NODES * NODES; idx += 128)
    rows[(idx >> 7) * 129 + (idx & 127)] = sigP[idx];
  s_active[i] = 1;
  __syncthreads();

  const float* myrow = &rows[i * 129];
  float m = NEGV;                                 // row max over active cols
  for (int j = 0; j < NODES; ++j) m = fmaxf(m, myrow[j]);

  for (int t = 0; t < NODES; ++t) {
    if (i == 0) { s_sel[0] = 1 << 30; s_sel[1] = -1; }
    const bool act = (s_active[i] != 0);
    float l = act ? (1.0f - m) : NEGV;

    // block max of logits: wave32 shfl tree + 4-way LDS combine
    float mx = l;
    for (int off = 16; off; off >>= 1) mx = fmaxf(mx, __shfl_xor(mx, off));
    if (lane == 0) s_red[wid] = mx;
    __syncthreads();
    mx = fmaxf(fmaxf(s_red[0], s_red[1]), fmaxf(s_red[2], s_red[3]));

    float e  = __expf(l - mx);                    // inactive -> exp(-1e30) = 0
    float ps = e;                                 // inclusive wave scan
    for (int off = 1; off < 32; off <<= 1) {
      float n = __shfl_up(ps, off);
      if (lane >= off) ps += n;
    }
    if (lane == 31) s_wsum[wid] = ps;
    __syncthreads();
    float base = 0.0f, total = 0.0f;
    for (int w = 0; w < 4; ++w) {
      float v = s_wsum[w];
      total += v;
      if (w < wid) base += v;
    }
    const float cum = (ps + base) / total;        // normalized cumsum
    const float u   = uniforms[d * NODES + t];

    if (act) {
      atomicMax(&s_sel[1], i);                    // last active (fallback)
      if (cum >= u) atomicMin(&s_sel[0], i);      // first satisfying index
    }
    __syncthreads();
    const int idx = (s_sel[0] < NODES) ? s_sel[0] : s_sel[1];
    if (i == idx) { s_order[t] = idx; s_active[idx] = 0; }
    __syncthreads();

    // incremental row-max: only rescan if removed column held the max
    if (myrow[idx] >= m) {
      m = NEGV;
      for (int j = 0; j < NODES; ++j)
        if (s_active[j]) m = fmaxf(m, myrow[j]);
    }
  }

  const int oi = s_order[i];
  order_ws[d * NODES + i]   = oi;
  inv_ws[d * NODES + oi]    = i;                  // inv[order[a]] = a
  order_out[d * NODES + i]  = (float)oi;          // second tuple output
}

// ---------------------------------------------------------------- Phase 2
__global__ void __launch_bounds__(256)
eb_sample_permute_kernel(const float* __restrict__ P,
                         const float* __restrict__ gu,
                         const int*   __restrict__ order_ws,
                         const int*   __restrict__ inv_ws,
                         float* __restrict__ out) {
  __shared__ _Float16 sWt[NODES * WSTR];          // sWt[c][k] = Y[inv[c], k]
  __shared__ int s_order[NODES];
  __shared__ int s_inv[NODES];

  const int d   = blockIdx.x;
  const int tid = threadIdx.x;

  if (tid < NODES) {
    s_order[tid] = order_ws[d * NODES + tid];
    s_inv[tid]   = inv_ws[d * NODES + tid];
  }
  __syncthreads();

  const float* g0p = gu + (size_t)d * 2u * NODES * NODES;
  const float* g1p = g0p + NODES * NODES;

  // 4 elements / thread / iter: float4 gumbel loads, ds_store_b64 to LDS.
  for (int p4 = tid * 4; p4 < NODES * NODES; p4 += 1024) {
    __builtin_prefetch(&g0p[p4 + 4096], 0, 3);    // global_prefetch_b8
    __builtin_prefetch(&g1p[p4 + 4096], 0, 3);
    const int a  = p4 >> 7;
    const int b0 = p4 & 127;
    const int oa = s_order[a];
    const v4f u0 = *(const v4f*)&g0p[p4];
    const v4f u1 = *(const v4f*)&g1p[p4];
    v4h yv;
    for (int e = 0; e < 4; ++e) {
      const int b = b0 + e;
      float y = 0.0f;
      if (a > b) {                                // strict lower triangle
        const float gg0 = -__logf(-__logf(u0[e] + 1e-10f) + 1e-10f);
        const float gg1 = -__logf(-__logf(u1[e] + 1e-10f) + 1e-10f);
        // logA + g0 >= log1mA + g1  <=>  P + (g0 - g1) >= 0
        const float w = P[oa * NODES + s_order[b]] + gg0 - gg1;
        y = (w >= 0.0f) ? 1.0f : 0.0f;
      }
      yv[e] = (_Float16)y;
    }
    *(v4h*)&sWt[oa * WSTR + b0] = yv;             // transposed+permuted store
  }
  __syncthreads();

  // out = Pm * W, one 16-row tile band per wave (8 waves = 128 rows)
  const int lane = tid & 31;
  const int wv   = tid >> 5;                      // tile-row mt
  const int r16  = lane & 15;
  const int hi   = lane >> 4;

  // A (=Pm) fragments for kt=0..3, contiguous k order matching B: k = hi*16 + e
  const int mrow = wv * 16 + r16;
  const int im   = s_inv[mrow];                   // Pm[m,k] = (k == inv[m])
  v16h afr[4];
  for (int kt = 0; kt < 4; ++kt) {
    v16h a;
    for (int e = 0; e < 16; ++e) {
      const int k = kt * 32 + (hi * 16 + e);
      a[e] = (k == im) ? (_Float16)1.0f : (_Float16)0.0f;
    }
    afr[kt] = a;
  }

  float* outd = out + (size_t)d * NODES * NODES;
  for (int nt = 0; nt < 8; ++nt) {
    const int c = nt * 16 + r16;                  // output column (lane-major)
    v8f acc = {};
    for (int kt = 0; kt < 4; ++kt) {
      // B layout: lane = column, k = kt*32 + hi*16 + e -> 16 contiguous halves
      const _Float16* bp = &sWt[c * WSTR + kt * 32 + hi * 16];
      const v8h blo = *(const v8h*)bp;            // ds_load_b128
      const v8h bhi = *(const v8h*)(bp + 8);      // ds_load_b128
      const v16h b = __builtin_shufflevector(blo, bhi,
          0, 1, 2, 3, 4, 5, 6, 7, 8, 9, 10, 11, 12, 13, 14, 15);
      acc = __builtin_amdgcn_wmma_f32_16x16x32_f16(
          false, afr[kt], false, b, (short)0, acc, false, false);
    }
    // C/D layout: VGPR g -> row m = g + hi*8
    for (int g = 0; g < 8; ++g) {
      const int mo = wv * 16 + hi * 8 + g;
      outd[mo * NODES + c] = acc[g];
    }
  }
}

// ---------------------------------------------------------------- launch
extern "C" void kernel_launch(void* const* d_in, const int* in_sizes, int n_in,
                              void* d_out, int out_size, void* d_ws, size_t ws_size,
                              hipStream_t stream) {
  (void)n_in; (void)out_size; (void)ws_size;
  const float* P        = (const float*)d_in[0];   // (128,128) f32
  const float* uniforms = (const float*)d_in[1];   // (D,128)   f32
  const float* gumbel_u = (const float*)d_in[2];   // (D,2,128,128) f32
  const int D = in_sizes[1] / NODES;               // 1024

  float* sigP     = (float*)d_ws;
  int*   order_ws = (int*)((char*)d_ws + (size_t)NODES * NODES * sizeof(float));
  int*   inv_ws   = order_ws + (size_t)D * NODES;

  float* out       = (float*)d_out;                      // (D,128,128)
  float* order_out = out + (size_t)D * NODES * NODES;    // (D,128) as f32

  eb_sigmoid_kernel<<<(NODES * NODES + 255) / 256, 256, 0, stream>>>(P, sigP);

  const size_t sh1 = (size_t)NODES * 129 * sizeof(float)  // sigP rows (padded)
                   + 2 * NODES * sizeof(int)              // order + active
                   + 8 * sizeof(float)                    // s_red + s_wsum
                   + 2 * sizeof(int);                     // s_sel
  eb_order_kernel<<<D, 128, sh1, stream>>>(sigP, uniforms, order_ws, inv_ws,
                                           order_out);

  eb_sample_permute_kernel<<<D, 256, 0, stream>>>(P, gumbel_u, order_ws,
                                                  inv_ws, out);
}
